// WinWinLayAttnProcessorSDXL_63282048139323
// MI455X (gfx1250) — matrix operand order repaired and checked by reading, not compiled, not run on hardware
//
#include <hip/hip_runtime.h>
#include <stdint.h>

#define DEV __device__ __forceinline__

typedef __attribute__((ext_vector_type(16))) __bf16        bfrag;   // A/B matrix frag (16 bf16 / lane)
typedef __attribute__((ext_vector_type(8)))  float         ffrag;   // C/D matrix frag (8 f32 / lane)
typedef __attribute__((ext_vector_type(4)))  unsigned int  u32x4;
typedef __attribute__((ext_vector_type(8)))  unsigned int  u32x8;
typedef __attribute__((ext_vector_type(2)))  unsigned int  u32x2;
typedef unsigned short bf16_t;  // bf16 storage

// Problem constants (SDXL down-block attention)
constexpr int BATCH = 2, SEQ = 4096, CH = 640, NH = 10, HD = 64;
constexpr int ROWS  = BATCH * SEQ;          // 8192
constexpr float SCALE = 0.125f;             // 1/sqrt(64), folded into Q

// ---------- CDNA5 async global->LDS path (probe-guarded) ----------
#if __has_builtin(__builtin_amdgcn_global_load_async_to_lds_b128)
#define HAVE_ASYNC_LDS 1
#endif

// Matches the builtin's parameter 0 type from the compiler diagnostic:
// '__attribute__((__vector_size__(4 * sizeof(int)))) int *'
typedef __attribute__((__vector_size__(4 * sizeof(int)))) int v4i;
typedef __attribute__((address_space(3))) v4i* lds_v4i_ptr;

DEV void copy16(const bf16_t* g, bf16_t* l) {
#if defined(HAVE_ASYNC_LDS)
  __builtin_amdgcn_global_load_async_to_lds_b128(
      (v4i*)g, (lds_v4i_ptr)(void*)l, 0, 0);
#else
  *reinterpret_cast<u32x4*>(l) = *reinterpret_cast<const u32x4*>(g);
#endif
}

DEV void wait_async_le2() {
#if defined(HAVE_ASYNC_LDS)
# if __has_builtin(__builtin_amdgcn_s_wait_asynccnt)
  __builtin_amdgcn_s_wait_asynccnt(2);
# else
  asm volatile("s_wait_asynccnt 0x2" ::: "memory");
# endif
#endif
}

DEV void wait_async_le0() {
#if defined(HAVE_ASYNC_LDS)
# if __has_builtin(__builtin_amdgcn_s_wait_asynccnt)
  __builtin_amdgcn_s_wait_asynccnt(0);
# else
  asm volatile("s_wait_asynccnt 0x0" ::: "memory");
# endif
#endif
}

// ---------- helpers ----------
DEV unsigned int bfbits(float f) {          // f32 -> bf16 bits, round-to-nearest-even
  unsigned int u = __builtin_bit_cast(unsigned int, f);
  u += 0x7FFFu + ((u >> 16) & 1u);
  return u >> 16;
}

DEV ffrag fzero() {
  ffrag z;
#pragma unroll
  for (int i = 0; i < 8; ++i) z[i] = 0.0f;
  return z;
}

// Build a 16-bf16 fragment from two contiguous 8-bf16 (16B) runs.
DEV bfrag load_frag(const bf16_t* p0, const bf16_t* p1) {
  u32x4 a = *reinterpret_cast<const u32x4*>(p0);
  u32x4 b = *reinterpret_cast<const u32x4*>(p1);
  u32x8 r;
  r[0] = a[0]; r[1] = a[1]; r[2] = a[2]; r[3] = a[3];
  r[4] = b[0]; r[5] = b[1]; r[6] = b[2]; r[7] = b[3];
  return __builtin_bit_cast(bfrag, r);
}

DEV ffrag wmma_bf16(bfrag a, bfrag b, ffrag c) {
  return __builtin_amdgcn_wmma_f32_16x16x32_bf16(false, a, false, b, (short)0, c, false, false);
}

// ---------- kernel 1a: hidden f32 -> bf16 (4 elems / thread) ----------
__global__ __launch_bounds__(256) void pack_bf16_x4(const float* __restrict__ in,
                                                    bf16_t* __restrict__ out, int n4) {
  int i = blockIdx.x * 256 + threadIdx.x;
  if (i >= n4) return;
  float4 v = reinterpret_cast<const float4*>(in)[i];
  u32x2 o;
  o[0] = bfbits(v.x) | (bfbits(v.y) << 16);
  o[1] = bfbits(v.z) | (bfbits(v.w) << 16);
  reinterpret_cast<u32x2*>(out)[i] = o;
}

// ---------- kernel 1b: W[k][n] f32 -> WT[n][k] bf16 ----------
__global__ __launch_bounds__(256) void transpose_bf16(const float* __restrict__ in,
                                                      bf16_t* __restrict__ out) {
  int i = blockIdx.x * 256 + threadIdx.x;   // CH*CH
  int n = i % CH, k = i / CH;               // coalesced reads over n
  out[(size_t)n * CH + k] = (bf16_t)bfbits(in[(size_t)k * CH + n]);
}

// ---------- kernel 2: QKV projection (one 16x16 tile per wave) ----------
// A-frag (16x32 bf16): lane L row=L%16; elem i -> k = kk + i + 16*(L/16)
// B-frag (32x16 bf16): lane L col=L%16; elem i -> k = kk + i + 16*(L/16)   (from WT row n)
__global__ __launch_bounds__(256) void qkv_gemm(const bf16_t* __restrict__ Hb,
                                                const bf16_t* __restrict__ WqT,
                                                const bf16_t* __restrict__ WkT,
                                                const bf16_t* __restrict__ WvT,
                                                bf16_t* __restrict__ Qb,
                                                bf16_t* __restrict__ Kb,
                                                bf16_t* __restrict__ Vt) {
  const int wave = threadIdx.x >> 5, lane = threadIdx.x & 31;
  const int lm = lane & 15, half = lane >> 4;
  int gid = blockIdx.x * 8 + wave;                 // 3 * 512 * 40 wave-tiles
  const int mat = gid / (512 * 40);
  int rem = gid % (512 * 40);
  const int mt = rem / 40, nt = rem % 40;
  const bf16_t* WT = (mat == 0) ? WqT : (mat == 1) ? WkT : WvT;

  const bf16_t* arow = Hb + (size_t)(mt * 16 + lm) * CH + 16 * half;
  const bf16_t* brow = WT + (size_t)(nt * 16 + lm) * CH + 16 * half;
  ffrag acc = fzero();
#pragma unroll 4
  for (int kk = 0; kk < CH; kk += 32) {
    bfrag a = load_frag(arow + kk, arow + kk + 8);
    bfrag b = load_frag(brow + kk, brow + kk + 8);
    acc = wmma_bf16(a, b, acc);
  }
  const int c = nt * 16 + lm, h = c >> 6, d = c & 63;
#pragma unroll
  for (int r = 0; r < 8; ++r) {
    int m = mt * 16 + r + 8 * half;                // C-frag: row r (+8 for hi half)
    size_t bh = (size_t)(m >> 12) * NH + h;
    int pos = m & 4095;
    float v = acc[r];
    if (mat == 0)      Qb[(bh * SEQ + pos) * HD + d] = (bf16_t)bfbits(v * SCALE);
    else if (mat == 1) Kb[(bh * SEQ + pos) * HD + d] = (bf16_t)bfbits(v);
    else               Vt[(bh * HD + d) * SEQ + pos] = (bf16_t)bfbits(v);
  }
}

// ---------- kernel 3: flash attention ----------
// Block = 8 waves; wave owns 16 query rows x D=64 f32 accumulators.
// K/V chunks (32 keys) staged once per block into LDS via async global->LDS,
// double-buffered: chunk kc+1 copies overlap compute on chunk kc.
__global__ __launch_bounds__(256) void attn_kernel(const bf16_t* __restrict__ Qb,
                                                   const bf16_t* __restrict__ Kb,
                                                   const bf16_t* __restrict__ Vt,
                                                   bf16_t* __restrict__ AO) {
  __shared__ alignas(16) bf16_t psh[8][16][32];     // per-wave P re-layout tile (8 KB)
  __shared__ alignas(16) bf16_t ktile[2][32][64];   // [buf][key][d]   2 x 4 KB
  __shared__ alignas(16) bf16_t vtile[2][64][32];   // [buf][d][key]   2 x 4 KB
  const int tid = threadIdx.x;
  const int wave = tid >> 5, lane = tid & 31;
  const int lm = lane & 15, half = lane >> 4;
  const int bh = blockIdx.x >> 5;                   // batch*head
  const int b = bh / NH, h = bh % NH;
  const int qbase = (blockIdx.x & 31) * 128 + wave * 16;

  const bf16_t* kh = Kb + (size_t)bh * SEQ * HD;
  const bf16_t* vh = Vt + (size_t)bh * HD * SEQ;

  // Each thread copies one 16B slice of K and one of V per chunk (2 async / thread).
  auto issue_chunk = [&](int kc, int buf) {
    copy16(kh + (size_t)kc * 32 * HD + tid * 8, &ktile[buf][0][0] + tid * 8);
    copy16(vh + (size_t)(tid >> 2) * SEQ + kc * 32 + (tid & 3) * 8,
           &vtile[buf][tid >> 2][(tid & 3) * 8]);
  };

  // Q A-frags for k-window 0..31 and 32..63 (Q pre-scaled by 1/sqrt(D))
  const bf16_t* qrow = Qb + ((size_t)bh * SEQ + qbase + lm) * HD + 16 * half;
  bfrag aq0 = load_frag(qrow, qrow + 8);
  bfrag aq1 = load_frag(qrow + 32, qrow + 40);

  ffrag o0 = fzero(), o1 = fzero(), o2 = fzero(), o3 = fzero();
  float mrun[8], lrun[8];
#pragma unroll
  for (int r = 0; r < 8; ++r) { mrun[r] = -3.0e38f; lrun[r] = 0.0f; }

  issue_chunk(0, 0);
  issue_chunk(1, 1);

  for (int kc = 0; kc < SEQ / 32; ++kc) {
    const int cur = kc & 1;
    if (kc + 1 < SEQ / 32) wait_async_le2();  // chunk kc landed; kc+1 may be in flight
    else                   wait_async_le0();
    __syncthreads();                          // all waves' slices visible

    // S = Q K^T for 32 keys: two 16x16 C-frags, 2 k-steps each (from LDS)
    const bf16_t* kr0 = &ktile[cur][lm][16 * half];
    const bf16_t* kr1 = &ktile[cur][16 + lm][16 * half];
    ffrag s0 = fzero(), s1 = fzero();
    s0 = wmma_bf16(aq0, load_frag(kr0, kr0 + 8), s0);
    s0 = wmma_bf16(aq1, load_frag(kr0 + 32, kr0 + 40), s0);
    s1 = wmma_bf16(aq0, load_frag(kr1, kr1 + 8), s1);
    s1 = wmma_bf16(aq1, load_frag(kr1 + 32, kr1 + 40), s1);

    // online softmax: per C-frag slot r -> row (r + 8*half), cols across 16 lanes
#pragma unroll
    for (int r = 0; r < 8; ++r) {
      float x = fmaxf(s0[r], s1[r]);
      x = fmaxf(x, __shfl_xor(x, 1));
      x = fmaxf(x, __shfl_xor(x, 2));
      x = fmaxf(x, __shfl_xor(x, 4));
      x = fmaxf(x, __shfl_xor(x, 8));
      float mnew = fmaxf(mrun[r], x);
      float alpha = __expf(mrun[r] - mnew);
      mrun[r] = mnew;
      float p0 = __expf(s0[r] - mnew);
      float p1 = __expf(s1[r] - mnew);
      float rs = p0 + p1;
      rs += __shfl_xor(rs, 1);
      rs += __shfl_xor(rs, 2);
      rs += __shfl_xor(rs, 4);
      rs += __shfl_xor(rs, 8);
      lrun[r] = lrun[r] * alpha + rs;
      o0[r] *= alpha; o1[r] *= alpha; o2[r] *= alpha; o3[r] *= alpha;
      psh[wave][r + 8 * half][lm]      = (bf16_t)bfbits(p0);
      psh[wave][r + 8 * half][16 + lm] = (bf16_t)bfbits(p1);
    }
    // reload P as A-frag (16x32); compiler inserts s_wait_dscnt for the RAW
    const bf16_t* prow = &psh[wave][lm][16 * half];
    bfrag pa = load_frag(prow, prow + 8);

    // O += P V : B-frags from LDS V tile (row = d, contiguous 16 keys)
    const bf16_t* vr = &vtile[cur][lm][16 * half];
    o0 = wmma_bf16(pa, load_frag(vr, vr + 8), o0);
    o1 = wmma_bf16(pa, load_frag(vr + 16 * 32, vr + 16 * 32 + 8), o1);
    o2 = wmma_bf16(pa, load_frag(vr + 32 * 32, vr + 32 * 32 + 8), o2);
    o3 = wmma_bf16(pa, load_frag(vr + 48 * 32, vr + 48 * 32 + 8), o3);

    __syncthreads();                          // everyone done reading tile[cur]
    if (kc + 2 < SEQ / 32) issue_chunk(kc + 2, cur);
  }

  // normalize and store to AO[b][n][h*64+d] (bf16)
  const size_t obase = ((size_t)b * SEQ + qbase) * CH + h * HD + lm;
#pragma unroll
  for (int r = 0; r < 8; ++r) {
    float inv = 1.0f / lrun[r];
    size_t ro = obase + (size_t)(r + 8 * half) * CH;
    AO[ro]      = (bf16_t)bfbits(o0[r] * inv);
    AO[ro + 16] = (bf16_t)bfbits(o1[r] * inv);
    AO[ro + 32] = (bf16_t)bfbits(o2[r] * inv);
    AO[ro + 48] = (bf16_t)bfbits(o3[r] * inv);
  }
}

// ---------- kernel 4: output projection + bias ----------
__global__ __launch_bounds__(256) void out_gemm(const bf16_t* __restrict__ AO,
                                                const bf16_t* __restrict__ WoT,
                                                const float* __restrict__ bo,
                                                float* __restrict__ out) {
  const int wave = threadIdx.x >> 5, lane = threadIdx.x & 31;
  const int lm = lane & 15, half = lane >> 4;
  int gid = blockIdx.x * 8 + wave;                 // 512*40 wave-tiles
  const int mt = gid / 40, nt = gid % 40;
  const bf16_t* arow = AO + (size_t)(mt * 16 + lm) * CH + 16 * half;
  const bf16_t* brow = WoT + (size_t)(nt * 16 + lm) * CH + 16 * half;
  ffrag acc = fzero();
#pragma unroll 4
  for (int kk = 0; kk < CH; kk += 32)
    acc = wmma_bf16(load_frag(arow + kk, arow + kk + 8),
                    load_frag(brow + kk, brow + kk + 8), acc);
  const int c = nt * 16 + lm;
  const float bias = bo[c];
#pragma unroll
  for (int r = 0; r < 8; ++r)
    out[(size_t)(mt * 16 + r + 8 * half) * CH + c] = acc[r] + bias;
}

// ---------- workspace layout (bytes) ----------
constexpr size_t HB_B  = (size_t)ROWS * CH * 2;              // 10.5 MB
constexpr size_t WT_B  = (size_t)CH * CH * 2;                // 0.8 MB
constexpr size_t OFF_WQT = HB_B;
constexpr size_t OFF_WKT = OFF_WQT + WT_B;
constexpr size_t OFF_WVT = OFF_WKT + WT_B;
constexpr size_t OFF_WOT = OFF_WVT + WT_B;
constexpr size_t OFF_QB  = OFF_WOT + WT_B;
constexpr size_t OFF_KB  = OFF_QB + HB_B;
constexpr size_t OFF_VT  = OFF_KB + HB_B;
constexpr size_t OFF_AO  = OFF_VT + HB_B;

extern "C" void kernel_launch(void* const* d_in, const int* in_sizes, int n_in,
                              void* d_out, int out_size, void* d_ws, size_t ws_size,
                              hipStream_t stream) {
  const float* hidden = (const float*)d_in[0];
  const float* Wq = (const float*)d_in[1];
  const float* Wk = (const float*)d_in[2];
  const float* Wv = (const float*)d_in[3];
  const float* Wo = (const float*)d_in[4];
  const float* bo = (const float*)d_in[5];
  float* out = (float*)d_out;
  char* ws = (char*)d_ws;

  bf16_t* Hb  = (bf16_t*)(ws);
  bf16_t* WqT = (bf16_t*)(ws + OFF_WQT);
  bf16_t* WkT = (bf16_t*)(ws + OFF_WKT);
  bf16_t* WvT = (bf16_t*)(ws + OFF_WVT);
  bf16_t* WoT = (bf16_t*)(ws + OFF_WOT);
  bf16_t* Qb  = (bf16_t*)(ws + OFF_QB);
  bf16_t* Kb  = (bf16_t*)(ws + OFF_KB);
  bf16_t* Vt  = (bf16_t*)(ws + OFF_VT);
  bf16_t* AO  = (bf16_t*)(ws + OFF_AO);

  pack_bf16_x4<<<ROWS * CH / 4 / 256, 256, 0, stream>>>(hidden, Hb, ROWS * CH / 4);
  transpose_bf16<<<CH * CH / 256, 256, 0, stream>>>(Wq, WqT);
  transpose_bf16<<<CH * CH / 256, 256, 0, stream>>>(Wk, WkT);
  transpose_bf16<<<CH * CH / 256, 256, 0, stream>>>(Wv, WvT);
  transpose_bf16<<<CH * CH / 256, 256, 0, stream>>>(Wo, WoT);
  qkv_gemm<<<3 * 512 * 40 / 8, 256, 0, stream>>>(Hb, WqT, WkT, WvT, Qb, Kb, Vt);
  attn_kernel<<<BATCH * NH * (SEQ / 128), 256, 0, stream>>>(Qb, Kb, Vt, AO);
  out_gemm<<<512 * 40 / 8, 256, 0, stream>>>(AO, WoT, bo, out);
}
